// TopologicalMap_80324478370286
// MI455X (gfx1250) — matrix operand order repaired and checked by reading, not compiled, not run on hardware
//
#include <hip/hip_runtime.h>
#include <hip/hip_bf16.h>

typedef __attribute__((ext_vector_type(16))) __bf16 v16bf;
typedef __attribute__((ext_vector_type(8)))  float  v8f;

#define GM_N 1024
#define GM_C 4096
#define GM_K 16384
#define SIDE 128

__device__ __forceinline__ unsigned short f2bf(float f) {
    unsigned int u = __float_as_uint(f);
    unsigned int r = u + 0x7FFFu + ((u >> 16) & 1u);   // round-to-nearest-even
    return (unsigned short)(r >> 16);
}

// Async DMA: global (16B per lane) -> LDS, tracked by ASYNCcnt (CDNA5 §08).
__device__ __forceinline__ void async_copy_b128(unsigned lds_off, const void* gptr) {
    asm volatile("global_load_async_to_lds_b128 %0, %1, off"
                 :: "v"(lds_off), "v"(gptr) : "memory");
}
__device__ __forceinline__ void wait_async0() {
    asm volatile("s_wait_asynccnt 0x0" ::: "memory");
}

// ---------------------------------------------------------------- x -> bf16, x2[n] = sum x^2
__global__ __launch_bounds__(256) void convx_kernel(const float* __restrict__ x,
                                                    unsigned short* __restrict__ xb,
                                                    float* __restrict__ x2) {
    int n = blockIdx.x, t = threadIdx.x;
    const float* xr = x + (size_t)n * GM_C;
    unsigned short* xbr = xb + (size_t)n * GM_C;
    float s = 0.0f;
    for (int c = t; c < GM_C; c += 256) {
        float v = xr[c];
        s += v * v;
        xbr[c] = f2bf(v);
    }
    __shared__ float red[256];
    red[t] = s; __syncthreads();
    for (int k = 128; k > 0; k >>= 1) {
        if (t < k) red[t] += red[t + k];
        __syncthreads();
    }
    if (t == 0) x2[n] = red[0];
}

// ---------------------------------------------------------------- w2[k] = sum_c W[c][k]^2
__global__ __launch_bounds__(256) void w2_kernel(const float* __restrict__ w,
                                                 float* __restrict__ w2) {
    int k = blockIdx.x * 256 + threadIdx.x;
    float s = 0.0f;
    for (int c = 0; c < GM_C; ++c) {
        float v = w[(size_t)c * GM_K + k];
        s += v * v;
    }
    w2[k] = s;
}

// ---------------------------------------------------------------- W (C,K) fp32 -> WT (K,C) bf16
__global__ __launch_bounds__(256) void wtr_kernel(const float* __restrict__ w,
                                                  unsigned short* __restrict__ wbT) {
    __shared__ unsigned short tile[64][72];
    int k0 = blockIdx.x * 64;   // K tile
    int c0 = blockIdx.y * 64;   // C tile
    int t = threadIdx.x;
#pragma unroll
    for (int i = 0; i < 16; ++i) {
        int idx = t + i * 256;
        int r = idx >> 6, cc = idx & 63;           // r along C, cc along K (coalesced read)
        tile[r][cc] = f2bf(w[(size_t)(c0 + r) * GM_K + (k0 + cc)]);
    }
    __syncthreads();
#pragma unroll
    for (int i = 0; i < 16; ++i) {
        int idx = t + i * 256;
        int rr = idx >> 6, c = idx & 63;           // rr along K, c along C (coalesced write)
        wbT[(size_t)(k0 + rr) * GM_C + (c0 + c)] = tile[c][rr];
    }
}

// ---------------------------------------------------------------- BF16 WMMA GEMM + norms2 epilogue
// Block tile 128x128, BK=32 (one v_wmma_f32_16x16x32_bf16 K-step).
// 8 waves in 2x4 grid; each wave owns 64x32 = 4x2 tiles of 16x16.
// Double-buffered LDS filled by GLOBAL_LOAD_ASYNC_TO_LDS_B128 (no VGPR staging).
union FragU {
    uint4 q[2];
    v16bf v;
};

__global__ __launch_bounds__(256) void gemm_kernel(const unsigned short* __restrict__ xb,
                                                   const unsigned short* __restrict__ wbT,
                                                   const float* __restrict__ x2,
                                                   const float* __restrict__ w2,
                                                   float* __restrict__ out) {
    constexpr int BK = 32;
    constexpr int LDA = 40;                 // ushorts: 32 + 8 pad -> 80B row stride, conflict-free
    constexpr int KT = GM_C / BK;           // 128 K-steps

    __shared__ __align__(16) unsigned short As[2][128 * LDA];
    __shared__ __align__(16) unsigned short Bs[2][128 * LDA];

    int t = threadIdx.x;
    int lane = t & 31, wid = t >> 5;
    int wm = wid >> 2, wn = wid & 3;        // 2x4 wave grid
    int half = lane >> 4, r16 = lane & 15;
    int m0 = blockIdx.y * 128;
    int n0 = blockIdx.x * 128;

    // Per-thread DMA chunks: 512 A-chunks + 512 B-chunks of 16B per k-step,
    // 256 threads -> 2 A + 2 B async issues per thread per k-step.
    int q0 = t, q1 = t + 256;
    int ra0 = q0 >> 2, ca0 = q0 & 3;
    int ra1 = q1 >> 2, ca1 = q1 & 3;
    const unsigned short* gA0 = xb  + (size_t)(m0 + ra0) * GM_C + ca0 * 8;
    const unsigned short* gA1 = xb  + (size_t)(m0 + ra1) * GM_C + ca1 * 8;
    const unsigned short* gB0 = wbT + (size_t)(n0 + ra0) * GM_C + ca0 * 8;
    const unsigned short* gB1 = wbT + (size_t)(n0 + ra1) * GM_C + ca1 * 8;
    unsigned lA0[2], lA1[2], lB0[2], lB1[2];
#pragma unroll
    for (int b = 0; b < 2; ++b) {
        lA0[b] = (unsigned)(uintptr_t)&As[b][ra0 * LDA + ca0 * 8];
        lA1[b] = (unsigned)(uintptr_t)&As[b][ra1 * LDA + ca1 * 8];
        lB0[b] = (unsigned)(uintptr_t)&Bs[b][ra0 * LDA + ca0 * 8];
        lB1[b] = (unsigned)(uintptr_t)&Bs[b][ra1 * LDA + ca1 * 8];
    }

    auto issue = [&](int buf, int kt) {
        int c0 = kt * BK;
        async_copy_b128(lA0[buf], gA0 + c0);
        async_copy_b128(lA1[buf], gA1 + c0);
        async_copy_b128(lB0[buf], gB0 + c0);
        async_copy_b128(lB1[buf], gB1 + c0);
    };

    v8f acc[4][2];
#pragma unroll
    for (int mt = 0; mt < 4; ++mt)
#pragma unroll
        for (int nt = 0; nt < 2; ++nt) acc[mt][nt] = (v8f){};

    issue(0, 0);
    wait_async0();
    __syncthreads();

#pragma unroll 1
    for (int kt = 0; kt < KT; ++kt) {
        int cur = kt & 1;
        if (kt + 1 < KT) issue(cur ^ 1, kt + 1);   // DMA next tile while computing

        // fragment loads (contiguous 16B chunks per ISA A/B lane layout)
        FragU fa[4], fb[2];
#pragma unroll
        for (int mt = 0; mt < 4; ++mt) {
            const unsigned short* ap = &As[cur][(wm * 64 + mt * 16 + r16) * LDA];
            fa[mt].q[0] = *(const uint4*)(ap + half * 8);
            fa[mt].q[1] = *(const uint4*)(ap + 16 + half * 8);
        }
#pragma unroll
        for (int nt = 0; nt < 2; ++nt) {
            const unsigned short* bp = &Bs[cur][(wn * 32 + nt * 16 + r16) * LDA];
            fb[nt].q[0] = *(const uint4*)(bp + half * 8);
            fb[nt].q[1] = *(const uint4*)(bp + 16 + half * 8);
        }
#pragma unroll
        for (int mt = 0; mt < 4; ++mt)
#pragma unroll
            for (int nt = 0; nt < 2; ++nt)
                acc[mt][nt] = __builtin_amdgcn_wmma_f32_16x16x32_bf16(
                    false, fa[mt].v, false, fb[nt].v, (short)0, acc[mt][nt], false, false);

        if (kt + 1 < KT) {
            wait_async0();       // our DMA into next buffer has landed in LDS
            __syncthreads();     // everyone's DMA landed; everyone done reading cur
        }
    }

    // epilogue: norms2 = max(x2 + w2 - 2*dot, 0)
#pragma unroll
    for (int mt = 0; mt < 4; ++mt)
#pragma unroll
        for (int nt = 0; nt < 2; ++nt)
#pragma unroll
            for (int i = 0; i < 8; ++i) {
                int gr = m0 + wm * 64 + mt * 16 + half * 8 + i;
                int gc = n0 + wn * 32 + nt * 16 + r16;
                float v = x2[gr] + w2[gc] - 2.0f * acc[mt][nt][i];
                out[(size_t)gr * GM_K + gc] = v > 0.0f ? v : 0.0f;
            }
}

// ---------------------------------------------------------------- argmin per row (first occurrence)
__global__ __launch_bounds__(256) void argmin_kernel(const float* __restrict__ out,
                                                     int* __restrict__ bmu) {
    int n = blockIdx.x, t = threadIdx.x;
    const float* row = out + (size_t)n * GM_K;
    float bv = 3.402823466e38f;
    int bi = GM_K;
    for (int k = t; k < GM_K; k += 256) {
        float v = row[k];
        if (v < bv) { bv = v; bi = k; }
    }
    __shared__ float sv[256];
    __shared__ int si[256];
    sv[t] = bv; si[t] = bi; __syncthreads();
    for (int s = 128; s > 0; s >>= 1) {
        if (t < s) {
            if (sv[t + s] < sv[t] || (sv[t + s] == sv[t] && si[t + s] < si[t])) {
                sv[t] = sv[t + s]; si[t] = si[t + s];
            }
        }
        __syncthreads();
    }
    if (t == 0) bmu[n] = si[0];
}

// ---------------------------------------------------------------- separable normalized Gaussian
__global__ __launch_bounds__(256) void phi_kernel(float* __restrict__ out,
                                                  const int* __restrict__ bmu,
                                                  const float* __restrict__ stdp) {
    int n = blockIdx.x, t = threadIdx.x;
    int b = bmu[n];
    float rowf = (float)(b >> 7);
    float colf = (float)(b & (SIDE - 1));
    float sd = stdp[0];
    float c = 0.5f / (sd * sd);

    __shared__ float er[SIDE], ec[SIDE];
    __shared__ float sums[2];
    if (t < SIDE) {
        float d = (float)t - rowf;
        er[t] = expf(-d * d * c);
    } else {
        int j = t - SIDE;
        float d = (float)j - colf;
        ec[j] = expf(-d * d * c);
    }
    __syncthreads();
    if (t == 0) { float s = 0.f; for (int i = 0; i < SIDE; ++i) s += er[i]; sums[0] = s; }
    if (t == 1) { float s = 0.f; for (int i = 0; i < SIDE; ++i) s += ec[i]; sums[1] = s; }
    __syncthreads();
    float inv = 1.0f / (sums[0] * sums[1]);
    float* row = out + (size_t)n * GM_K;
    for (int k = t; k < GM_K; k += 256) {
        int gi = k >> 7, gj = k & (SIDE - 1);
        row[k] = row[k] * (er[gi] * ec[gj] * inv);
    }
}

// ----------------------------------------------------------------
extern "C" void kernel_launch(void* const* d_in, const int* in_sizes, int n_in,
                              void* d_out, int out_size, void* d_ws, size_t ws_size,
                              hipStream_t stream) {
    const float* x    = (const float*)d_in[0];   // (N, C)
    const float* w    = (const float*)d_in[1];   // (C, K)
    const float* stdp = (const float*)d_in[2];   // scalar
    float* out = (float*)d_out;                  // (N, K)

    char* ws = (char*)d_ws;
    size_t off = 0;
    unsigned short* xb  = (unsigned short*)(ws + off); off += (size_t)GM_N * GM_C * 2;  // 8 MB
    unsigned short* wbT = (unsigned short*)(ws + off); off += (size_t)GM_K * GM_C * 2;  // 128 MB
    float* x2 = (float*)(ws + off); off += (size_t)GM_N * 4;
    float* w2 = (float*)(ws + off); off += (size_t)GM_K * 4;
    int* bmu  = (int*)(ws + off);   off += (size_t)GM_N * 4;
    (void)ws_size; (void)in_sizes; (void)n_in; (void)out_size;

    convx_kernel<<<GM_N, 256, 0, stream>>>(x, xb, x2);
    w2_kernel<<<GM_K / 256, 256, 0, stream>>>(w, w2);
    wtr_kernel<<<dim3(GM_K / 64, GM_C / 64), 256, 0, stream>>>(w, wbT);
    gemm_kernel<<<dim3(GM_K / 128, GM_N / 128), 256, 0, stream>>>(xb, wbT, x2, w2, out);
    argmin_kernel<<<GM_N, 256, 0, stream>>>(out, bmu);
    phi_kernel<<<GM_N, 256, 0, stream>>>(out, bmu, stdp);
}
